// GraphTransformer_20005957664874
// MI455X (gfx1250) — matrix operand order repaired and checked by reading, not compiled, not run on hardware
//
#include <hip/hip_runtime.h>
#include <math.h>

#define DIMC   256
#define DEPTHC 2
#define HEADSC 8
#define DHC    64
#define INNERC 512
#define EDIMC  256
#define BC     2
#define NC     256
#define FFDC   1024
#define BNC    (BC*NC)
#define EPSC   1e-5f
#define FMAXC  3.402823466e38f

typedef __attribute__((ext_vector_type(16))) __bf16 bf16x16;
typedef __attribute__((ext_vector_type(8)))  __bf16 bf16x8;
typedef __attribute__((ext_vector_type(8)))  float  f32x8;

static __device__ __forceinline__ __bf16 f2bf(float f) {
    unsigned u = __builtin_bit_cast(unsigned, f);
    u += 0x7fffu + ((u >> 16) & 1u);          // round-to-nearest-even
    unsigned short h = (unsigned short)(u >> 16);
    return __builtin_bit_cast(__bf16, h);
}
static __device__ __forceinline__ float bf2f(__bf16 b) {
    unsigned short h = __builtin_bit_cast(unsigned short, b);
    unsigned u = ((unsigned)h) << 16;
    return __builtin_bit_cast(float, u);
}
static __device__ __forceinline__ float wred(float v) {
#pragma unroll
    for (int off = 16; off > 0; off >>= 1) v += __shfl_xor(v, off, 32);
    return v;
}

// A-fragment for v_wmma_*_16x16x32_bf16: lane L holds row L&15;
// idx 0..7  = K[k0 + (L>>4)*8 .. +7], idx 8..15 = K[k0+16 + (L>>4)*8 .. +7]
static __device__ __forceinline__ bf16x16 load_a_frag(const __bf16* Abase, int lda,
                                                      int k0, int lane) {
    const __bf16* pr = Abase + (size_t)(lane & 15) * lda;
    int half = lane >> 4;
    bf16x8 lo = *(const bf16x8*)(pr + k0 + half * 8);
    bf16x8 hi = *(const bf16x8*)(pr + k0 + 16 + half * 8);
    bf16x16 a;
#pragma unroll
    for (int i = 0; i < 8; i++) { a[i] = lo[i]; a[i + 8] = hi[i]; }
    return a;
}

// ---------------- edge layernorm: f32 out (final output) + bf16 copy (L2-resident GEMM operand)
__global__ void edge_ln_kernel(const float* __restrict__ edges,
                               const float* __restrict__ g,
                               const float* __restrict__ bln,
                               float* __restrict__ outf,
                               __bf16* __restrict__ outb) {
    int wave = threadIdx.x >> 5, lane = threadIdx.x & 31;
    size_t row = (size_t)blockIdx.x * 8 + wave;
    const float* p = edges + row * EDIMC;
    float x[8]; float s = 0.f, s2 = 0.f;
#pragma unroll
    for (int k = 0; k < 8; k++) { x[k] = p[lane + 32 * k]; s += x[k]; s2 += x[k] * x[k]; }
    s = wred(s); s2 = wred(s2);
    float m = s * (1.f / EDIMC);
    float var = s2 * (1.f / EDIMC) - m * m;
    float r = rsqrtf(var + EPSC);
#pragma unroll
    for (int k = 0; k < 8; k++) {
        int c = lane + 32 * k;
        float y = (x[k] - m) * r * g[c] + bln[c];
        outf[row * EDIMC + c] = y;
        outb[row * EDIMC + c] = f2bf(y);
    }
}

// ---------------- node layernorm -> bf16 activations
__global__ void node_ln_kernel(const float* __restrict__ nodes,
                               const float* __restrict__ g,
                               const float* __restrict__ bln,
                               __bf16* __restrict__ outb) {
    int wave = threadIdx.x >> 5, lane = threadIdx.x & 31;
    size_t row = (size_t)blockIdx.x * 8 + wave;
    const float* p = nodes + row * DIMC;
    float x[8]; float s = 0.f, s2 = 0.f;
#pragma unroll
    for (int k = 0; k < 8; k++) { x[k] = p[lane + 32 * k]; s += x[k]; s2 += x[k] * x[k]; }
    s = wred(s); s2 = wred(s2);
    float m = s * (1.f / DIMC);
    float var = s2 * (1.f / DIMC) - m * m;
    float r = rsqrtf(var + EPSC);
#pragma unroll
    for (int k = 0; k < 8; k++) {
        int c = lane + 32 * k;
        outb[row * DIMC + c] = f2bf((x[k] - m) * r * g[c] + bln[c]);
    }
}

// ---------------- pack f32 [K,N] weight into WMMA B-fragment layout:
// packed[nt][ks][lane][16] ; lane = col(L&15); idx 0..7 = K[(L>>4)*8 ..+7], idx 8..15 = K[16+(L>>4)*8 ..+7]
__global__ void pack_b_kernel(const float* __restrict__ Bw, int K, int N,
                              __bf16* __restrict__ out) {
    int gid = blockIdx.x * blockDim.x + threadIdx.x;
    int lane = gid & 31, tile = gid >> 5;
    int Ksteps = K >> 5;
    int ks = tile % Ksteps, nt = tile / Ksteps;
    int col = nt * 16 + (lane & 15);
    int kbase = ks * 32 + (lane >> 4) * 8;
    __bf16* o = out + (size_t)gid * 16;
#pragma unroll
    for (int i = 0; i < 16; i++) o[i] = f2bf(Bw[(size_t)(kbase + (i & 7) + (i >> 3) * 16) * N + col]);
}

// ---------------- generic WMMA GEMM: C[M,N] = A_bf16[M,K] @ Bpacked + bias ; act 1 = exact GELU
__global__ void gemm_kernel(const __bf16* __restrict__ A, int lda,
                            const __bf16* __restrict__ Bp,
                            int Ntiles, int Ksteps,
                            const float* __restrict__ bias,
                            float* __restrict__ Cf, __bf16* __restrict__ Cb,
                            int ldc, int act) {
    int wave = threadIdx.x >> 5, lane = threadIdx.x & 31;
    int m0 = blockIdx.x * 16;
    const __bf16* Abase = A + (size_t)m0 * lda;
    int half = lane >> 4, col = lane & 15;
    for (int nt = wave; nt < Ntiles; nt += 8) {
        const __bf16* Bt = Bp + (size_t)nt * Ksteps * 512;
        float bv = bias ? bias[nt * 16 + col] : 0.f;
        f32x8 c;
#pragma unroll
        for (int r = 0; r < 8; r++) c[r] = bv;
        for (int ks = 0; ks < Ksteps; ks++) {
            bf16x16 a = load_a_frag(Abase, lda, ks * 32, lane);
            bf16x16 b = *(const bf16x16*)(Bt + (size_t)ks * 512 + lane * 16);
            c = __builtin_amdgcn_wmma_f32_16x16x32_bf16(false, a, false, b,
                                                        (short)0, c, false, false);
        }
#pragma unroll
        for (int r = 0; r < 8; r++) {
            int row = m0 + r + 8 * half;
            float v = c[r];
            if (act == 1) v = 0.5f * v * (1.f + erff(v * 0.70710678118654752f));
            int cc = nt * 16 + col;
            if (Cf) Cf[(size_t)row * ldc + cc] = v;
            if (Cb) Cb[(size_t)row * ldc + cc] = f2bf(v);
        }
    }
}

// ---------------- WMMA GEMM (N=256 fixed) fused with sigmoid gated residual; updates nodes in place
__global__ void gemm_gated_kernel(const __bf16* __restrict__ A, int lda,
                                  const __bf16* __restrict__ Bp, int Ksteps,
                                  const float* __restrict__ bias,
                                  const float* __restrict__ grw,
                                  float* __restrict__ nodes) {
    __shared__ float c_s[16][DIMC];
    __shared__ float part_s[16][16];
    __shared__ float g_s[16];
    int wave = threadIdx.x >> 5, lane = threadIdx.x & 31;
    int m0 = blockIdx.x * 16;
    const __bf16* Abase = A + (size_t)m0 * lda;
    int half = lane >> 4, col = lane & 15;
    for (int nt = wave; nt < 16; nt += 8) {
        const __bf16* Bt = Bp + (size_t)nt * Ksteps * 512;
        float bv = bias[nt * 16 + col];
        f32x8 c;
#pragma unroll
        for (int r = 0; r < 8; r++) c[r] = bv;
        for (int ks = 0; ks < Ksteps; ks++) {
            bf16x16 a = load_a_frag(Abase, lda, ks * 32, lane);
            bf16x16 b = *(const bf16x16*)(Bt + (size_t)ks * 512 + lane * 16);
            c = __builtin_amdgcn_wmma_f32_16x16x32_bf16(false, a, false, b,
                                                        (short)0, c, false, false);
        }
#pragma unroll
        for (int r = 0; r < 8; r++) c_s[r + 8 * half][nt * 16 + col] = c[r];
    }
    __syncthreads();
    int rrow = threadIdx.x >> 4, seg = threadIdx.x & 15;
    float* rowp = nodes + (size_t)(m0 + rrow) * DIMC;
    float p = 0.f;
#pragma unroll
    for (int i = 0; i < 16; i++) {
        int d = seg * 16 + i;
        float xv = c_s[rrow][d], rv = rowp[d];
        p += xv * grw[d] + rv * grw[DIMC + d] + (xv - rv) * grw[2 * DIMC + d];
    }
    part_s[rrow][seg] = p;
    __syncthreads();
    if (seg == 0) {
        float t = 0.f;
#pragma unroll
        for (int i = 0; i < 16; i++) t += part_s[rrow][i];
        g_s[rrow] = 1.f / (1.f + expf(-t));
    }
    __syncthreads();
    float gg = g_s[rrow];
#pragma unroll
    for (int i = 0; i < 16; i++) {
        int d = seg * 16 + i;
        float xv = c_s[rrow][d], rv = rowp[d];
        rowp[d] = xv * gg + rv * (1.f - gg);   // each element owned by exactly one thread
    }
}

// ---------------- fused attention: per (b, head, 16-row i-tile); flash-style over j-tiles.
// e-projection via WMMA from L2-resident bf16 edges; score/value contraction in f32 VALU.
__global__ void attn_kernel(const __bf16* __restrict__ edgesb,
                            const __bf16* __restrict__ WePack,
                            const float* __restrict__ be,
                            const float* __restrict__ qbuf,
                            const float* __restrict__ kvbuf,
                            const unsigned char* __restrict__ mask,
                            __bf16* __restrict__ attn_out) {
    __shared__ __bf16 e_s[16][16][DHC];     // 32 KB
    __shared__ float  q_s[16][DHC];
    __shared__ float  k_s[16][DHC];
    __shared__ float  v_s[16][DHC];
    __shared__ float  o_s[16][DHC];
    __shared__ float  sim_s[16][16];
    __shared__ float  row_m[16], row_l[16];

    int tid = threadIdx.x;
    int wave = tid >> 5, lane = tid & 31;
    int it = blockIdx.x & 15;
    int h  = (blockIdx.x >> 4) & 7;
    int b  = blockIdx.x >> 7;
    int i0 = it * 16;
    const float scale = 0.125f;              // DH^-0.5
    int pi = tid >> 4, dg = tid & 15;
    int half = lane >> 4, col = lane & 15;

#pragma unroll
    for (int u = 0; u < 4; u++) {
        int d = dg * 4 + u;
        q_s[pi][d] = qbuf[((size_t)(b * NC + i0 + pi)) * INNERC + h * DHC + d] * scale;
        o_s[pi][d] = 0.f;
    }
    if (tid < 16) { row_m[tid] = -FMAXC; row_l[tid] = 0.f; }

    for (int jt = 0; jt < 16; jt++) {
        int j0 = jt * 16;
        __syncthreads();
        // stage K/V tiles (f32)
#pragma unroll
        for (int u = 0; u < 4; u++) {
            int d = dg * 4 + u;
            size_t base = ((size_t)(b * NC + j0 + pi)) * (2 * INNERC) + h * DHC + d;
            k_s[pi][d] = kvbuf[base];
            v_s[pi][d] = kvbuf[base + INNERC];
        }
        // e-projection GEMM: each wave does 2 i-rows x 4 d-tiles; K=256 (8 WMMA steps)
#pragma unroll
        for (int mi = 0; mi < 2; mi++) {
            int mt = wave * 2 + mi;          // i index within tile
            const __bf16* Abase = edgesb + (((size_t)(b * NC + i0 + mt)) * NC + j0) * EDIMC;
            if (jt + 1 < 16)                 // pull next j-tile toward L2/L0
                __builtin_prefetch(Abase + 16 * EDIMC, 0, 1);
#pragma unroll
            for (int nt = 0; nt < 4; nt++) {
                float bv = be[h * DHC + nt * 16 + col];
                f32x8 c;
#pragma unroll
                for (int r = 0; r < 8; r++) c[r] = bv;
                const __bf16* Bt = WePack + (size_t)(h * 4 + nt) * 8 * 512;
#pragma unroll
                for (int ks = 0; ks < 8; ks++) {
                    bf16x16 a = load_a_frag(Abase, EDIMC, ks * 32, lane);
                    bf16x16 bb = *(const bf16x16*)(Bt + (size_t)ks * 512 + lane * 16);
                    c = __builtin_amdgcn_wmma_f32_16x16x32_bf16(false, a, false, bb,
                                                                (short)0, c, false, false);
                }
#pragma unroll
                for (int r = 0; r < 8; r++)
                    e_s[mt][r + 8 * half][nt * 16 + col] = f2bf(c[r]);
            }
        }
        __syncthreads();
        // scores: one (i,j) element per thread
        {
            int pj = dg;
            float s = 0.f;
#pragma unroll
            for (int d = 0; d < DHC; d++)
                s += q_s[pi][d] * (k_s[pj][d] + bf2f(e_s[pi][pj][d]));
            bool mm = mask[b * NC + i0 + pi] && mask[b * NC + j0 + pj];
            sim_s[pi][pj] = mm ? s : -FMAXC;
        }
        __syncthreads();
        // online softmax update; thread owns (row pi, 4 dims)
        float old_m = row_m[pi], old_l = row_l[pi];
        float nm = old_m;
#pragma unroll
        for (int pj = 0; pj < 16; pj++) nm = fmaxf(nm, sim_s[pi][pj]);
        float factor = expf(old_m - nm);
        float pv[16]; float ladd = 0.f;
#pragma unroll
        for (int pj = 0; pj < 16; pj++) { pv[pj] = expf(sim_s[pi][pj] - nm); ladd += pv[pj]; }
#pragma unroll
        for (int u = 0; u < 4; u++) {
            int d = dg * 4 + u;
            float acc = o_s[pi][d] * factor;
#pragma unroll
            for (int pj = 0; pj < 16; pj++)
                acc += pv[pj] * (v_s[pj][d] + bf2f(e_s[pi][pj][d]));
            o_s[pi][d] = acc;
        }
        __syncthreads();
        if (dg == 0) { row_m[pi] = nm; row_l[pi] = old_l * factor + ladd; }
    }
    __syncthreads();
    float inv = 1.f / row_l[pi];
#pragma unroll
    for (int u = 0; u < 4; u++) {
        int d = dg * 4 + u;
        attn_out[((size_t)(b * NC + i0 + pi)) * INNERC + h * DHC + d] = f2bf(o_s[pi][d] * inv);
    }
}

extern "C" void kernel_launch(void* const* d_in, const int* in_sizes, int n_in,
                              void* d_out, int out_size, void* d_ws, size_t ws_size,
                              hipStream_t stream) {
    const float* nodes_in = (const float*)d_in[0];
    const float* edges    = (const float*)d_in[1];
    const unsigned char* mask = (const unsigned char*)d_in[2];
    const float* eg   = (const float*)d_in[3];
    const float* ebln = (const float*)d_in[4];
    const float* ln1g = (const float*)d_in[5];
    const float* ln1b = (const float*)d_in[6];
    const float* Wq   = (const float*)d_in[7];
    const float* bq   = (const float*)d_in[8];
    const float* Wkv  = (const float*)d_in[9];
    const float* bkv  = (const float*)d_in[10];
    const float* We   = (const float*)d_in[11];
    const float* be   = (const float*)d_in[12];
    const float* Wo   = (const float*)d_in[13];
    const float* bo   = (const float*)d_in[14];
    const float* gr1  = (const float*)d_in[15];
    const float* ln2g = (const float*)d_in[16];
    const float* ln2b = (const float*)d_in[17];
    const float* W1   = (const float*)d_in[18];
    const float* b1   = (const float*)d_in[19];
    const float* W2   = (const float*)d_in[20];
    const float* b2   = (const float*)d_in[21];
    const float* gr2  = (const float*)d_in[22];

    float* out_nodes = (float*)d_out;
    float* out_edges = out_nodes + (size_t)BNC * DIMC;

    char* ws = (char*)d_ws;
    __bf16* edgesb = (__bf16*)ws; ws += (size_t)BC * NC * NC * EDIMC * 2;   // 67 MB, L2-resident
    __bf16* pWq  = (__bf16*)ws; ws += (size_t)DIMC * INNERC * 2;
    __bf16* pWkv = (__bf16*)ws; ws += (size_t)DIMC * 2 * INNERC * 2;
    __bf16* pWe  = (__bf16*)ws; ws += (size_t)EDIMC * INNERC * 2;
    __bf16* pWo  = (__bf16*)ws; ws += (size_t)INNERC * DIMC * 2;
    __bf16* pW1  = (__bf16*)ws; ws += (size_t)DIMC * FFDC * 2;
    __bf16* pW2  = (__bf16*)ws; ws += (size_t)FFDC * DIMC * 2;
    __bf16* xln  = (__bf16*)ws; ws += (size_t)BNC * DIMC * 2;
    float*  qbuf = (float*)ws;  ws += (size_t)BNC * INNERC * 4;
    float*  kvbuf= (float*)ws;  ws += (size_t)BNC * 2 * INNERC * 4;
    __bf16* aout = (__bf16*)ws; ws += (size_t)BNC * INNERC * 2;
    __bf16* hbuf = (__bf16*)ws; ws += (size_t)BNC * FFDC * 2;

    // living node state = first output region
    hipMemcpyAsync(out_nodes, nodes_in, (size_t)BNC * DIMC * 4,
                   hipMemcpyDeviceToDevice, stream);
    edge_ln_kernel<<<BC * NC * NC / 8, 256, 0, stream>>>(edges, eg, ebln, out_edges, edgesb);

    for (int l = 0; l < DEPTHC; l++) {
        auto pack = [&](const float* Bsrc, int K, int N, __bf16* dst) {
            int threads = (N / 16) * (K / 32) * 32;
            pack_b_kernel<<<threads / 256, 256, 0, stream>>>(Bsrc, K, N, dst);
        };
        pack(Wq  + (size_t)l * DIMC * INNERC,     DIMC,  INNERC,     pWq);
        pack(Wkv + (size_t)l * DIMC * 2 * INNERC, DIMC,  2 * INNERC, pWkv);
        pack(We  + (size_t)l * EDIMC * INNERC,    EDIMC, INNERC,     pWe);
        pack(Wo  + (size_t)l * INNERC * DIMC,     INNERC, DIMC,      pWo);
        pack(W1  + (size_t)l * DIMC * FFDC,       DIMC,  FFDC,       pW1);
        pack(W2  + (size_t)l * FFDC * DIMC,       FFDC,  DIMC,       pW2);

        node_ln_kernel<<<BNC / 8, 256, 0, stream>>>(out_nodes, ln1g + l * DIMC,
                                                    ln1b + l * DIMC, xln);
        gemm_kernel<<<BNC / 16, 256, 0, stream>>>(xln, DIMC, pWq, INNERC / 16, DIMC / 32,
                                                  bq + (size_t)l * INNERC, qbuf, nullptr,
                                                  INNERC, 0);
        gemm_kernel<<<BNC / 16, 256, 0, stream>>>(xln, DIMC, pWkv, 2 * INNERC / 16, DIMC / 32,
                                                  bkv + (size_t)l * 2 * INNERC, kvbuf, nullptr,
                                                  2 * INNERC, 0);
        attn_kernel<<<BC * HEADSC * (NC / 16), 256, 0, stream>>>(edgesb, pWe,
                                                  be + (size_t)l * INNERC, qbuf, kvbuf,
                                                  mask, aout);
        gemm_gated_kernel<<<BNC / 16, 256, 0, stream>>>(aout, INNERC, pWo, INNERC / 32,
                                                  bo + (size_t)l * DIMC,
                                                  gr1 + (size_t)l * 3 * DIMC, out_nodes);
        node_ln_kernel<<<BNC / 8, 256, 0, stream>>>(out_nodes, ln2g + l * DIMC,
                                                    ln2b + l * DIMC, xln);
        gemm_kernel<<<BNC / 16, 256, 0, stream>>>(xln, DIMC, pW1, FFDC / 16, DIMC / 32,
                                                  b1 + (size_t)l * FFDC, nullptr, hbuf,
                                                  FFDC, 1);
        gemm_gated_kernel<<<BNC / 16, 256, 0, stream>>>(hbuf, FFDC, pW2, FFDC / 32,
                                                  b2 + (size_t)l * DIMC,
                                                  gr2 + (size_t)l * 3 * DIMC, out_nodes);
    }
}
